// OpticalAttention_68934225101139
// MI455X (gfx1250) — matrix operand order repaired and checked by reading, not compiled, not run on hardware
//
#include <hip/hip_runtime.h>
#include <hip/hip_bf16.h>

typedef __attribute__((ext_vector_type(16))) _Float16 v16h;
typedef __attribute__((ext_vector_type(8)))  _Float16 v8h;
typedef __attribute__((ext_vector_type(8)))  float    v8f;
typedef __attribute__((ext_vector_type(8)))  float    f32x8;

constexpr int Eg = 1024;   // embed dim
constexpr int Hh = 16;     // heads
constexpr int Dd = 64;     // head dim
constexpr int Tt = 2048;   // seq len
constexpr int Bb = 4;      // batch
constexpr int Mrows = Bb * Tt;  // 8192

// base-2 softmax: exp(x/sqrt(D)) == exp2(x * log2(e)/sqrt(D))
constexpr float SCALE_L2 = 0.125f * 1.44269504088896f;

// P-tile LDS row stride in halves (16x64 tile + 8 pad: 144B = 36 banks,
// 36*r mod 64 distinct for r=0..15 -> conflict-free row access).
constexpr int PSTR = 72;

// ---------------------------------------------------------------------------
// WMMA helper: D = A(16x32 f16) x B(32x16 f16) + C(16x16 f32)
// ---------------------------------------------------------------------------
__device__ inline v8f wmma16(v16h a, v16h b, v8f c) {
  return __builtin_amdgcn_wmma_f32_16x16x32_f16(
      /*neg_a=*/false, a, /*neg_b=*/false, b,
      /*c_mod=*/(short)0, c, /*reuse_a=*/false, /*reuse_b=*/false);
}

// A-fragment: lane holds row M=lane%16,
// halves 0..7 -> K = base..base+7 ; halves 8..15 -> K = base+8..base+15,
// where base = (lane>>4)*16.  Two 16B loads per fragment.
__device__ inline v16h packA(v8h lo, v8h hi) {
  v16h a;
#pragma unroll
  for (int i = 0; i < 8; ++i) { a[i] = lo[i]; a[i + 8] = hi[i]; }
  return a;
}
__device__ inline v16h loadA_h(const _Float16* rowp, int base) {
  v8h lo = *reinterpret_cast<const v8h*>(rowp + base);
  v8h hi = *reinterpret_cast<const v8h*>(rowp + base + 8);
  return packA(lo, hi);
}
__device__ inline v16h loadA_f(const float* rowp, int base) {
  f32x8 lo = *reinterpret_cast<const f32x8*>(rowp + base);
  f32x8 hi = *reinterpret_cast<const f32x8*>(rowp + base + 8);
  v16h a;
#pragma unroll
  for (int i = 0; i < 8; ++i) {
    a[i] = (_Float16)lo[i];
    a[i + 8] = (_Float16)hi[i];
  }
  return a;
}
// B-fragment (this code's k order, same as loadA): lane holds column N=lane%16 with 16
// CONTIGUOUS K values starting at 16*(lane>>4).  One 32B load.
__device__ inline v16h loadB_h(const _Float16* p) {
  return *reinterpret_cast<const v16h*>(p);
}

// Row max across the 16 lanes of each wave-half (xor masks 1..8 never cross
// lane bit 4, so both halves reduce independently).
__device__ inline float rmax16(float v) {
#pragma unroll
  for (int m = 8; m >= 1; m >>= 1) v = fmaxf(v, __shfl_xor(v, m, 32));
  return v;
}

// ---------------------------------------------------------------------------
// fp32 -> fp16 convert (weights)
// ---------------------------------------------------------------------------
__global__ void cvt_f32_f16(const float* __restrict__ s, _Float16* __restrict__ d, int n) {
  int i = blockIdx.x * blockDim.x + threadIdx.x;
  int stride = gridDim.x * blockDim.x;
  for (; i < n; i += stride) d[i] = (_Float16)s[i];
}

// ---------------------------------------------------------------------------
// NT GEMM:  C[m,n] = sum_k A[m,k] * W[n,k] + bias[n]     (y = x W^T + b)
// MODE 0: A=f32, out=f16 at Q/K layout  [b,h,t,d]
// MODE 1: A=f32, out=f16 at V^T layout  [b,h,d,s]
// MODE 2: A=f16, out=f32 row-major      [m,n]
// Grid: (M/128, N/64), 256 thr (8 waves); each wave -> 16(M) x 64(N) tile.
// ---------------------------------------------------------------------------
template <int MODE>
__global__ __launch_bounds__(256) void gemm_nt(const void* __restrict__ Ain,
                                               const _Float16* __restrict__ W,
                                               const float* __restrict__ bias,
                                               void* __restrict__ Out) {
  constexpr bool AF32 = (MODE != 2);
  const int lane = threadIdx.x & 31;
  const int wave = threadIdx.x >> 5;
  const int row_in = lane & 15;
  const int hs = lane >> 4;
  const int baseA = hs * 16;
  const int m0 = blockIdx.x * 128 + wave * 16;
  const int n0 = blockIdx.y * 64;
  const int arow = m0 + row_in;

  v8f acc[4] = {};
  const float* Af = (const float*)Ain;
  const _Float16* Ah = (const _Float16*)Ain;

#pragma unroll 2
  for (int k0 = 0; k0 < Eg; k0 += 32) {
    v16h af;
    if constexpr (AF32)
      af = loadA_f(Af + (size_t)arow * Eg + k0, baseA);
    else
      af = loadA_h(Ah + (size_t)arow * Eg + k0, baseA);
#pragma unroll
    for (int j = 0; j < 4; ++j) {
      const _Float16* wp = W + (size_t)(n0 + j * 16 + row_in) * Eg + k0 + 16 * hs;
      acc[j] = wmma16(af, loadB_h(wp), acc[j]);
    }
  }

#pragma unroll
  for (int j = 0; j < 4; ++j) {
    const int n = n0 + j * 16 + row_in;
    const float bv = bias[n];
#pragma unroll
    for (int r = 0; r < 8; ++r) {
      const int m = m0 + r + 8 * hs;
      const float val = acc[j][r] + bv;
      if constexpr (MODE == 0) {
        const int b = m >> 11, t = m & (Tt - 1);
        const int head = n >> 6, d = n & 63;
        ((_Float16*)Out)[(((size_t)(b * Hh + head)) * Tt + t) * Dd + d] = (_Float16)val;
      } else if constexpr (MODE == 1) {
        const int b = m >> 11, s = m & (Tt - 1);
        const int head = n >> 6, d = n & 63;
        ((_Float16*)Out)[(((size_t)(b * Hh + head)) * Dd + d) * Tt + s] = (_Float16)val;
      } else {
        ((float*)Out)[(size_t)m * Eg + n] = val;
      }
    }
  }
}

// ---------------------------------------------------------------------------
// One 64-column flash-attention step for a 16-row wave tile.
// MASKED=false: strictly below the causal diagonal (no compares / selects).
// Per step: 8 score WMMA + 2 row-sum WMMA + 8 PV WMMA = 18 WMMA.
// ---------------------------------------------------------------------------
template <bool MASKED>
__device__ __forceinline__ void attn_step(
    int s0, int t0, int row_in, int hs, int baseA,
    const _Float16* __restrict__ kk, const _Float16* __restrict__ vt,
    _Float16* __restrict__ lp, const v16h (&qf)[2], const v16h& onesf,
    v8f (&o)[4], v8f& osum, float (&mrun)[8]) {
  // ---- scores: four 16x16 subtiles over s, contraction D=64 (2 WMMA each)
  v8f sc[4] = {};
#pragma unroll
  for (int st = 0; st < 4; ++st) {
    const _Float16* kr = kk + (size_t)(s0 + st * 16 + row_in) * Dd + 16 * hs;
    sc[st] = wmma16(qf[0], loadB_h(kr), sc[st]);
    sc[st] = wmma16(qf[1], loadB_h(kr + 32), sc[st]);
  }
  // ---- online softmax (C-layout: value (t=t0+r+8*hs, s=s0+16*st+lane%16))
#pragma unroll
  for (int r = 0; r < 8; ++r) {
    float v[4];
#pragma unroll
    for (int st = 0; st < 4; ++st) {
      v[st] = sc[st][r] * SCALE_L2;
      if constexpr (MASKED) {
        const int t = t0 + r + 8 * hs;
        const int s = s0 + st * 16 + row_in;
        v[st] = (s <= t) ? v[st] : -3.0e38f;
      }
    }
    const float tm = rmax16(fmaxf(fmaxf(v[0], v[1]), fmaxf(v[2], v[3])));
    const float nm = fmaxf(mrun[r], tm);
    const float corr = exp2f(mrun[r] - nm);  // v_exp_f32, no ln2 mul
    mrun[r] = nm;
    osum[r] *= corr;
#pragma unroll
    for (int j = 0; j < 4; ++j) o[j][r] *= corr;
    const int rr = r + 8 * hs;
#pragma unroll
    for (int st = 0; st < 4; ++st)
      lp[rr * PSTR + st * 16 + row_in] = (_Float16)exp2f(v[st] - nm);
  }
  // Same-wave LDS is in-order; fences only stop compiler reordering.
  asm volatile("" ::: "memory");
  const v16h pf0 = loadA_h(lp + row_in * PSTR, baseA);
  const v16h pf1 = loadA_h(lp + row_in * PSTR + 32, baseA);
  asm volatile("" ::: "memory");
  // ---- row sums on the matrix core: osum += P x ones
  osum = wmma16(pf0, onesf, osum);
  osum = wmma16(pf1, onesf, osum);
  // ---- O += P(16x64) x V(64x64): 8 WMMA, B-frags contiguous from V^T
#pragma unroll
  for (int j = 0; j < 4; ++j) {
    const _Float16* vr = vt + (size_t)(j * 16 + row_in) * Tt + s0 + 16 * hs;
    o[j] = wmma16(pf0, loadB_h(vr), o[j]);
    o[j] = wmma16(pf1, loadB_h(vr + 32), o[j]);
  }
}

// ---------------------------------------------------------------------------
// Causal flash attention.  Q,K: [B,H,T,D] f16.  VT: [B,H,D,S] f16.
// AO: [B,T,H,D] f16 (== row-major [B*T, E]).
// 8 waves / block; wave owns 16 query rows; online base-2 softmax; row sums
// via P x ones WMMA; 64-column steps; steady-state loop unmasked, short
// masked tail at the causal diagonal.
// ---------------------------------------------------------------------------
__global__ __launch_bounds__(256) void attn_fwd(const _Float16* __restrict__ Q,
                                                const _Float16* __restrict__ Kc,
                                                const _Float16* __restrict__ VT,
                                                _Float16* __restrict__ AO) {
  __shared__ _Float16 ldsP[8][16 * PSTR];
  const int lane = threadIdx.x & 31;
  const int wave = threadIdx.x >> 5;
  const int row_in = lane & 15;
  const int hs = lane >> 4;
  const int baseA = hs * 16;
  constexpr int TCH = Tt / 128;  // 16 row-chunks per (b,h)
  const int bh = blockIdx.x / TCH;
  const int t0 = (blockIdx.x % TCH) * 128 + wave * 16;
  const int b = bh >> 4, head = bh & 15;

  const _Float16* q = Q + (size_t)bh * Tt * Dd;
  const _Float16* kk = Kc + (size_t)bh * Tt * Dd;
  const _Float16* vt = VT + (size_t)bh * Dd * Tt;
  _Float16* lp = ldsP[wave];

  // Preload the two Q A-fragments (d 0..31 and 32..63) for this wave's rows.
  v16h qf[2];
  {
    const _Float16* qr = q + (size_t)(t0 + row_in) * Dd;
    qf[0] = loadA_h(qr, baseA);
    qf[1] = loadA_h(qr + 32, baseA);
  }

  // Constant ones B-fragment for the row-sum WMMA.
  v16h onesf;
#pragma unroll
  for (int i = 0; i < 16; ++i) onesf[i] = (_Float16)1.0f;

  v8f o[4] = {};
  v8f osum = {};                 // row-sum accumulator (every column = sum)
  float mrun[8];
#pragma unroll
  for (int r = 0; r < 8; ++r) mrun[r] = -3.0e38f;

  const int smax = t0 + 15;
  int s0 = 0;
  // Steady state: tile strictly below the diagonal -> no causal compares.
  for (; s0 + 63 <= t0; s0 += 64)
    attn_step<false>(s0, t0, row_in, hs, baseA, kk, vt, lp, qf, onesf, o, osum, mrun);
  // Diagonal tail: apply causal mask.
  for (; s0 <= smax; s0 += 64)
    attn_step<true>(s0, t0, row_in, hs, baseA, kk, vt, lp, qf, onesf, o, osum, mrun);

  // ---- epilogue: normalize and store f16 at [b, t, head, d]
#pragma unroll
  for (int r = 0; r < 8; ++r) {
    const float inv = 1.0f / osum[r];
    const int t = t0 + r + 8 * hs;
    _Float16* orow = AO + ((size_t)(b * Tt + t)) * Eg + head * Dd;
#pragma unroll
    for (int j = 0; j < 4; ++j)
      orow[j * 16 + row_in] = (_Float16)(o[j][r] * inv);
  }
}

// ---------------------------------------------------------------------------
extern "C" void kernel_launch(void* const* d_in, const int* in_sizes, int n_in,
                              void* d_out, int out_size, void* d_ws, size_t ws_size,
                              hipStream_t stream) {
  const float* query = (const float*)d_in[0];
  const float* key   = (const float*)d_in[1];
  const float* value = (const float*)d_in[2];
  const float* Wq = (const float*)d_in[3];
  const float* bq = (const float*)d_in[4];
  const float* Wk = (const float*)d_in[5];
  const float* bk = (const float*)d_in[6];
  const float* Wv = (const float*)d_in[7];
  const float* bv = (const float*)d_in[8];
  const float* Wo = (const float*)d_in[9];
  const float* bo = (const float*)d_in[10];

  _Float16* ws = (_Float16*)d_ws;
  const size_t WSZ = (size_t)Eg * Eg;    // 1M halves per weight matrix
  const size_t XSZ = (size_t)Mrows * Eg; // 8M halves per activation tensor
  _Float16* wqh = ws;
  _Float16* wkh = wqh + WSZ;
  _Float16* wvh = wkh + WSZ;
  _Float16* woh = wvh + WSZ;
  _Float16* qws = woh + WSZ;   // Q  [B,H,T,D]
  _Float16* kws = qws + XSZ;   // K  [B,H,S,D]
  _Float16* vws = kws + XSZ;   // V^T[B,H,D,S]
  _Float16* aows = vws + XSZ;  // attn out [B*T, E]

  cvt_f32_f16<<<1024, 256, 0, stream>>>(Wq, wqh, (int)WSZ);
  cvt_f32_f16<<<1024, 256, 0, stream>>>(Wk, wkh, (int)WSZ);
  cvt_f32_f16<<<1024, 256, 0, stream>>>(Wv, wvh, (int)WSZ);
  cvt_f32_f16<<<1024, 256, 0, stream>>>(Wo, woh, (int)WSZ);

  dim3 gg(Mrows / 128, Eg / 64);
  gemm_nt<0><<<gg, 256, 0, stream>>>(query, wqh, bq, qws);
  gemm_nt<0><<<gg, 256, 0, stream>>>(key,   wkh, bk, kws);
  gemm_nt<1><<<gg, 256, 0, stream>>>(value, wvh, bv, vws);

  attn_fwd<<<Bb * Hh * (Tt / 128), 256, 0, stream>>>(qws, kws, vws, aows);

  gemm_nt<2><<<gg, 256, 0, stream>>>(aows, woh, bo, d_out);
}
